// MambaBlock_50826642981200
// MI455X (gfx1250) — matrix operand order repaired and checked by reading, not compiled, not run on hardware
//
#include <hip/hip_runtime.h>

#define L_SEQ   2048
#define DMODEL  1024
#define DINNER  2048
#define DSTATE  16

typedef __attribute__((ext_vector_type(8)))  float  v8f;
typedef __attribute__((ext_vector_type(16))) __bf16 v16bf;
typedef __attribute__((ext_vector_type(8)))  __bf16 v8bf;

__device__ __forceinline__ float wave_sum(float v) {
    for (int off = 16; off >= 1; off >>= 1) v += __shfl_xor(v, off, 32);
    return v;
}

// ---- fp32 <-> bf16 split helpers (RNE, bit-exact, no scalar __bf16 math) ----
__device__ __forceinline__ unsigned short f2bf_rne(float f) {
    unsigned int u = __float_as_uint(f);
    u += 0x7FFFu + ((u >> 16) & 1u);
    return (unsigned short)(u >> 16);
}
__device__ __forceinline__ float bf2f(unsigned short h) {
    return __uint_as_float(((unsigned int)h) << 16);
}
__device__ __forceinline__ void split2(float x, unsigned short& hi, unsigned short& lo) {
    hi = f2bf_rne(x);
    lo = f2bf_rne(x - bf2f(hi));
}

// ---------------- LayerNorm: one block (256 thr) per row; emits hi/lo bf16 ----
__global__ __launch_bounds__(256) void ln_kernel(const float* __restrict__ x,
                                                 const float* __restrict__ g,
                                                 const float* __restrict__ b,
                                                 unsigned short* __restrict__ xn_hi,
                                                 unsigned short* __restrict__ xn_lo) {
    const int l = blockIdx.x, tid = threadIdx.x;
    const float* row = x + (size_t)l * DMODEL;
    float v[4], s = 0.f, s2 = 0.f;
    #pragma unroll
    for (int i = 0; i < 4; ++i) {
        v[i] = row[tid + i * 256];
        s += v[i]; s2 += v[i] * v[i];
    }
    s = wave_sum(s); s2 = wave_sum(s2);
    __shared__ float ss[8], ss2[8];
    const int wid = tid >> 5, lane = tid & 31;
    if (lane == 0) { ss[wid] = s; ss2[wid] = s2; }
    __syncthreads();
    float S = 0.f, S2 = 0.f;
    #pragma unroll
    for (int i = 0; i < 8; ++i) { S += ss[i]; S2 += ss2[i]; }
    const float mu   = S * (1.f / DMODEL);
    const float var  = S2 * (1.f / DMODEL) - mu * mu;
    const float rstd = rsqrtf(var + 1e-5f);
    #pragma unroll
    for (int i = 0; i < 4; ++i) {
        const int c = tid + i * 256;
        const float val = (v[i] - mu) * rstd * g[c] + b[c];
        unsigned short h, lo;
        split2(val, h, lo);
        xn_hi[(size_t)l * DMODEL + c] = h;
        xn_lo[(size_t)l * DMODEL + c] = lo;
    }
}

// ---------------- A = -exp(A_log) ----------------
__global__ void aprep_kernel(const float* __restrict__ A_log, float* __restrict__ Aneg) {
    const int i = blockIdx.x * 256 + threadIdx.x;
    Aneg[i] = -expf(A_log[i]);
}

// ---------------- weight split: f32 -> (hi, lo) bf16 ----------------
__global__ void split_kernel(const float* __restrict__ in,
                             unsigned short* __restrict__ hi,
                             unsigned short* __restrict__ lo, int n) {
    const int i = blockIdx.x * 256 + threadIdx.x;
    if (i < n) {
        unsigned short h, l;
        split2(in[i], h, l);
        hi[i] = h; lo[i] = l;
    }
}

// ---------------- WMMA fragment loads (bf16) ----------------
// A (16x32): lane m=lane&15, half=lane>>4 holds K = half*16 + {0..15} (contiguous)
__device__ __forceinline__ v16bf load_afrag(const unsigned short* p) {
    union { v16bf v; v8bf h[2]; } u;
    u.h[0] = *(const v8bf*)(p);
    u.h[1] = *(const v8bf*)(p + 8);
    return u.v;
}
// B (32x16): lane n=lane&15, half=lane>>4 holds K = half*16 + {0..15} (contiguous)
__device__ __forceinline__ v16bf load_bfrag(const unsigned short* p) {
    return *(const v16bf*)p;
}

#define WMMA_BF16(A, B, C) \
    __builtin_amdgcn_wmma_f32_16x16x32_bf16(false, (A), false, (B), (short)0, (C), false, false)

// ---------------- bf16x3 GEMM: out[M,N] = X[M,K] * W[N,K]^T (+ addend) ----------
// X ~ Xhi + Xlo, W ~ Whi + Wlo; out = Xhi*Whi + Xhi*Wlo + Xlo*Whi (fp32 acc).
// Block: 256 thr = 8 waves (2x4). Wave tile 32x32 = 2x2 WMMA tiles. Block: 64x128.
__global__ __launch_bounds__(256) void wmma_gemm_bf16x3(
        const unsigned short* __restrict__ Xhi, const unsigned short* __restrict__ Xlo,
        const unsigned short* __restrict__ Whi, const unsigned short* __restrict__ Wlo,
        float* __restrict__ out, const float* __restrict__ addend,
        int M, int N, int K) {
    const int wid  = threadIdx.x >> 5, lane = threadIdx.x & 31;
    const int wm   = wid >> 2, wn = wid & 3;
    const int row0 = blockIdx.y * 64 + wm * 32;
    const int col0 = blockIdx.x * 128 + wn * 32;
    const int mr   = lane & 15, half = lane >> 4;

    const size_t aoff = (size_t)(row0 + mr) * K + half * 16;
    const unsigned short* a0h = Xhi + aoff;
    const unsigned short* a0l = Xlo + aoff;
    const unsigned short* a1h = a0h + (size_t)16 * K;
    const unsigned short* a1l = a0l + (size_t)16 * K;
    const size_t boff = (size_t)(col0 + mr) * K + half * 16;
    const unsigned short* b0h = Whi + boff;
    const unsigned short* b0l = Wlo + boff;
    const unsigned short* b1h = b0h + (size_t)16 * K;
    const unsigned short* b1l = b0l + (size_t)16 * K;

    v8f c00 = {}, c01 = {}, c10 = {}, c11 = {};
    if (addend) {
        #pragma unroll
        for (int r = 0; r < 8; ++r) {
            const int rA = row0 + half * 8 + r;   // C/D layout: VGPR r -> M = r + half*8
            const int rB = rA + 16;
            c00[r] = addend[(size_t)rA * N + col0 + mr];
            c01[r] = addend[(size_t)rA * N + col0 + 16 + mr];
            c10[r] = addend[(size_t)rB * N + col0 + mr];
            c11[r] = addend[(size_t)rB * N + col0 + 16 + mr];
        }
    }

    #pragma unroll 2
    for (int k = 0; k < K; k += 32) {
        const v16bf A0h = load_afrag(a0h + k);
        const v16bf A0l = load_afrag(a0l + k);
        const v16bf A1h = load_afrag(a1h + k);
        const v16bf A1l = load_afrag(a1l + k);
        const v16bf B0h = load_bfrag(b0h + k);
        const v16bf B0l = load_bfrag(b0l + k);
        const v16bf B1h = load_bfrag(b1h + k);
        const v16bf B1l = load_bfrag(b1l + k);

        c00 = WMMA_BF16(A0h, B0h, c00);
        c00 = WMMA_BF16(A0h, B0l, c00);
        c00 = WMMA_BF16(A0l, B0h, c00);

        c01 = WMMA_BF16(A0h, B1h, c01);
        c01 = WMMA_BF16(A0h, B1l, c01);
        c01 = WMMA_BF16(A0l, B1h, c01);

        c10 = WMMA_BF16(A1h, B0h, c10);
        c10 = WMMA_BF16(A1h, B0l, c10);
        c10 = WMMA_BF16(A1l, B0h, c10);

        c11 = WMMA_BF16(A1h, B1h, c11);
        c11 = WMMA_BF16(A1h, B1l, c11);
        c11 = WMMA_BF16(A1l, B1h, c11);
    }

    #pragma unroll
    for (int r = 0; r < 8; ++r) {
        const int rA = row0 + half * 8 + r;
        const int rB = rA + 16;
        out[(size_t)rA * N + col0 + mr]      = c00[r];
        out[(size_t)rA * N + col0 + 16 + mr] = c01[r];
        out[(size_t)rB * N + col0 + mr]      = c10[r];
        out[(size_t)rB * N + col0 + 16 + mr] = c11[r];
    }
}

// ---------------- depthwise causal conv1d (k=4) + bias + SiLU ----------------
__global__ void conv_silu_kernel(const float* __restrict__ xz,   // (L, 2*DINNER)
                                 const float* __restrict__ cw,   // (DINNER,1,4)
                                 const float* __restrict__ cb,
                                 float* __restrict__ xc) {
    const int idx = blockIdx.x * 256 + threadIdx.x;   // L*DINNER
    const int d = idx % DINNER, l = idx / DINNER;
    float acc = cb[d];
    #pragma unroll
    for (int j = 0; j < 4; ++j) {
        const int t = l - 3 + j;
        if (t >= 0) acc += cw[d * 4 + j] * xz[(size_t)t * (2 * DINNER) + d];
    }
    xc[idx] = acc / (1.f + expf(-acc));   // silu
}

// ---------------- x_proj: (L,DINNER) @ (33,DINNER)^T -> dt, B, C ----------------
__global__ __launch_bounds__(256) void xproj_kernel(const float* __restrict__ xc,
                                                    const float* __restrict__ xw,
                                                    float* __restrict__ dtv,
                                                    float* __restrict__ Bm,
                                                    float* __restrict__ Cm) {
    __shared__ float xrow[DINNER];
    const int l = blockIdx.x, tid = threadIdx.x;
    for (int i = tid; i < DINNER; i += 256) xrow[i] = xc[(size_t)l * DINNER + i];
    __syncthreads();
    const int wid = tid >> 5, lane = tid & 31;
    for (int e = wid; e < 2 * DSTATE + 1; e += 8) {
        const float* w = xw + (size_t)e * DINNER;
        float s = 0.f;
        for (int i = lane; i < DINNER; i += 32) s += xrow[i] * w[i];
        s = wave_sum(s);
        if (lane == 0) {
            if (e == 0)            dtv[l] = s;
            else if (e <= DSTATE)  Bm[l * DSTATE + (e - 1)] = s;
            else                   Cm[l * DSTATE + (e - 1 - DSTATE)] = s;
        }
    }
}

// ------- selective scan: thread per (d,n); 16 lanes share d; emits hi/lo bf16 ---
__global__ __launch_bounds__(256) void scan_kernel(const float* __restrict__ xc,
                                                   const float* __restrict__ xz,
                                                   const float* __restrict__ dtv,
                                                   const float* __restrict__ Bm,
                                                   const float* __restrict__ Cm,
                                                   const float* __restrict__ Aneg,
                                                   const float* __restrict__ dt_w,
                                                   const float* __restrict__ dt_b,
                                                   const float* __restrict__ Dp,
                                                   unsigned short* __restrict__ y_hi,
                                                   unsigned short* __restrict__ y_lo) {
    const int gid = blockIdx.x * 256 + threadIdx.x;   // DINNER*DSTATE
    const int d = gid >> 4, n = gid & 15;
    const float A  = Aneg[d * DSTATE + n];
    const float w  = dt_w[d], bb = dt_b[d], Dd = Dp[d];
    float h = 0.f;
    for (int t = 0; t < L_SEQ; ++t) {
        const float v     = dtv[t] * w + bb;
        const float delta = (v > 20.f) ? v : log1pf(expf(v));    // softplus
        const float xcv   = xc[(size_t)t * DINNER + d];
        const float bv    = Bm[t * DSTATE + n];
        const float cv    = Cm[t * DSTATE + n];
        h = expf(delta * A) * h + delta * bv * xcv;
        float y = h * cv;
        #pragma unroll
        for (int off = 1; off < 16; off <<= 1) y += __shfl_xor(y, off, 32);
        if (n == 0) {
            const float z   = xz[(size_t)t * (2 * DINNER) + DINNER + d];
            const float sil = z / (1.f + expf(-z));
            const float out = (y + Dd * xcv) * sil;
            unsigned short hh, ll;
            split2(out, hh, ll);
            y_hi[(size_t)t * DINNER + d] = hh;
            y_lo[(size_t)t * DINNER + d] = ll;
        }
    }
}

extern "C" void kernel_launch(void* const* d_in, const int* in_sizes, int n_in,
                              void* d_out, int out_size, void* d_ws, size_t ws_size,
                              hipStream_t stream) {
    const float* x          = (const float*)d_in[0];
    const float* in_proj_w  = (const float*)d_in[1];
    const float* conv_w     = (const float*)d_in[2];
    const float* conv_b     = (const float*)d_in[3];
    const float* x_proj_w   = (const float*)d_in[4];
    const float* dt_w       = (const float*)d_in[5];
    const float* dt_b       = (const float*)d_in[6];
    const float* A_log      = (const float*)d_in[7];
    const float* Dp         = (const float*)d_in[8];
    const float* out_proj_w = (const float*)d_in[9];
    const float* ln_g       = (const float*)d_in[10];
    const float* ln_b       = (const float*)d_in[11];

    char* base = (char*)d_ws;
    auto alloc = [&](size_t bytes) { char* p = base; base += bytes; return p; };

    float* xz   = (float*)alloc((size_t)L_SEQ * 2 * DINNER * 4);   // 32 MB
    float* xc   = (float*)alloc((size_t)L_SEQ * DINNER * 4);       // 16 MB
    float* dtv  = (float*)alloc((size_t)L_SEQ * 4);
    float* Bm   = (float*)alloc((size_t)L_SEQ * DSTATE * 4);
    float* Cm   = (float*)alloc((size_t)L_SEQ * DSTATE * 4);
    float* Aneg = (float*)alloc((size_t)DINNER * DSTATE * 4);
    unsigned short* xn_hi = (unsigned short*)alloc((size_t)L_SEQ * DMODEL * 2);
    unsigned short* xn_lo = (unsigned short*)alloc((size_t)L_SEQ * DMODEL * 2);
    unsigned short* y_hi  = (unsigned short*)alloc((size_t)L_SEQ * DINNER * 2);
    unsigned short* y_lo  = (unsigned short*)alloc((size_t)L_SEQ * DINNER * 2);
    unsigned short* w1_hi = (unsigned short*)alloc((size_t)2 * DINNER * DMODEL * 2);
    unsigned short* w1_lo = (unsigned short*)alloc((size_t)2 * DINNER * DMODEL * 2);
    unsigned short* w2_hi = (unsigned short*)alloc((size_t)DMODEL * DINNER * 2);
    unsigned short* w2_lo = (unsigned short*)alloc((size_t)DMODEL * DINNER * 2);

    const int n_w1 = 2 * DINNER * DMODEL;   // 4M
    const int n_w2 = DMODEL * DINNER;       // 2M

    ln_kernel<<<L_SEQ, 256, 0, stream>>>(x, ln_g, ln_b, xn_hi, xn_lo);
    aprep_kernel<<<(DINNER * DSTATE) / 256, 256, 0, stream>>>(A_log, Aneg);
    split_kernel<<<(n_w1 + 255) / 256, 256, 0, stream>>>(in_proj_w, w1_hi, w1_lo, n_w1);
    split_kernel<<<(n_w2 + 255) / 256, 256, 0, stream>>>(out_proj_w, w2_hi, w2_lo, n_w2);

    // in_proj: (2048x1024) x (1024x4096) -> xz (f32)
    wmma_gemm_bf16x3<<<dim3((2 * DINNER) / 128, L_SEQ / 64), 256, 0, stream>>>(
        xn_hi, xn_lo, w1_hi, w1_lo, xz, nullptr, L_SEQ, 2 * DINNER, DMODEL);

    conv_silu_kernel<<<(L_SEQ * DINNER) / 256, 256, 0, stream>>>(xz, conv_w, conv_b, xc);
    xproj_kernel<<<L_SEQ, 256, 0, stream>>>(xc, x_proj_w, dtv, Bm, Cm);
    scan_kernel<<<(DINNER * DSTATE) / 256, 256, 0, stream>>>(
        xc, xz, dtv, Bm, Cm, Aneg, dt_w, dt_b, Dp, y_hi, y_lo);

    // out_proj + residual: (2048x2048) x (2048x1024) + x -> d_out
    wmma_gemm_bf16x3<<<dim3(DMODEL / 128, L_SEQ / 64), 256, 0, stream>>>(
        y_hi, y_lo, w2_hi, w2_lo, (float*)d_out, x, L_SEQ, DMODEL, DINNER);
}
